// MultiHeadSelfAttention_14345190768994
// MI455X (gfx1250) — compile-verified, matrix operand order checked by reading, not run on hardware
//
#include <hip/hip_runtime.h>
#include <hip/hip_bf16.h>

// ---------------------------------------------------------------------------
// MultiHeadSelfAttention over 3D volumes, MI455X (gfx1250) version.
//   B=2, C=256, H=8, P=32, X=Y=Z=48, N=110592 voxels.
// Pipeline:
//   1. weights f32 -> bf16
//   2. x [b,c,n] f32 -> xT [b,n,c] bf16   (LDS tile transpose)
//   3. q/k/v projections: WMMA bf16 GEMM, transposed bf16 output [b,n,o]
//   4. scores[b,hq,hk,x,y] = sum_{p,z} q*k / sqrt(P)
//   5. softmax over y (in place)
//   6. oT[b,n, h*32+c] = sum_m attn * vT
//   7. o-projection: WMMA bf16 GEMM, fp32 output in reference layout
// ---------------------------------------------------------------------------

typedef __bf16 bf16_t;
typedef __attribute__((ext_vector_type(8)))  __bf16 v8bf;
typedef __attribute__((ext_vector_type(16))) __bf16 v16bf;
typedef __attribute__((ext_vector_type(8)))  float  v8f;

#define C_DIM 256
#define NHEAD 8
#define PDIM  32
#define XDIM  48
#define NVOX  (48 * 48 * 48)   // 110592
#define BATCH 2

union FragBF { v16bf v; v8bf h[2]; };
union PackBF { v8bf v; __bf16 e[8]; };

// ---------------------------------------------------------------------------
// 1. plain f32 -> bf16 conversion (weights)
// ---------------------------------------------------------------------------
__global__ __launch_bounds__(256) void cvt_f32_bf16_kernel(
    const float* __restrict__ in, bf16_t* __restrict__ out, int n)
{
    int i = blockIdx.x * blockDim.x + threadIdx.x;
    if (i < n) out[i] = (__bf16)in[i];
}

// ---------------------------------------------------------------------------
// 2. transpose [b, C, N] f32 -> [b, N, C] bf16 via 32x32 LDS tiles
//    block (32,8), grid (N/32, C/32, B)
// ---------------------------------------------------------------------------
__global__ __launch_bounds__(256) void transpose_to_bf16_kernel(
    const float* __restrict__ x, bf16_t* __restrict__ xT)
{
    __shared__ float tile[32][33];
    const int b  = blockIdx.z;
    const int n0 = blockIdx.x * 32;
    const int c0 = blockIdx.y * 32;
    const int tx = threadIdx.x;      // 0..31
    const int ty = threadIdx.y;      // 0..7
    const float* xb = x + (size_t)b * C_DIM * NVOX;
#pragma unroll
    for (int i = 0; i < 32; i += 8)
        tile[ty + i][tx] = xb[(size_t)(c0 + ty + i) * NVOX + (n0 + tx)];
    __syncthreads();
    bf16_t* xTb = xT + (size_t)b * NVOX * C_DIM;
#pragma unroll
    for (int i = 0; i < 32; i += 8)
        xTb[(size_t)(n0 + ty + i) * C_DIM + (c0 + tx)] = (__bf16)tile[tx][ty + i];
}

// ---------------------------------------------------------------------------
// 3./7. WMMA bf16 GEMM:  out = W[256x256] * Bt^T  (+ bias)
//   W  : [256, 256] bf16 row-major (K contiguous)        -> A operand
//   Bt : [BATCH, NVOX, 256] bf16 (K contiguous per voxel) -> B operand
//   STORE_F32 == false : outT [BATCH, NVOX, 256] bf16 (transposed store)
//   STORE_F32 == true  : outN [BATCH, 256, NVOX] f32  (reference layout)
// block = 256 threads (8 waves); block tile 128(M) x 128(N);
// wave tile 64(M) x 32(N) = 4x2 WMMA tiles; K loop 8 x 32.
// ---------------------------------------------------------------------------
template <bool STORE_F32>
__global__ __launch_bounds__(256) void gemm256_wmma_kernel(
    const bf16_t* __restrict__ W,
    const bf16_t* __restrict__ Bt,
    const float*  __restrict__ bias,
    bf16_t* __restrict__ outT,
    float*  __restrict__ outN)
{
    const int bidx = blockIdx.z;
    const int lane = threadIdx.x & 31;
    const int wave = threadIdx.x >> 5;        // 0..7
    const int l15  = lane & 15;
    const int h    = lane >> 4;               // half-wave select
    const int m0   = blockIdx.y * 128 + (wave >> 2) * 64;
    const int n0   = blockIdx.x * 128 + (wave & 3) * 32;

    const bf16_t* Bb = Bt + (size_t)bidx * NVOX * C_DIM;

    // Per-lane fragment base pointers (ISA 16-bit A layout: two 8-elem K runs
    // at k0+8h and k0+16+8h for lane half h).
    const bf16_t* ap[4];
    const bf16_t* bp[2];
#pragma unroll
    for (int mi = 0; mi < 4; ++mi)
        ap[mi] = W + (size_t)(m0 + mi * 16 + l15) * C_DIM + 8 * h;
#pragma unroll
    for (int ni = 0; ni < 2; ++ni)
        bp[ni] = Bb + (size_t)(n0 + ni * 16 + l15) * C_DIM + 8 * h;

    v8f acc[4][2] = {};

    for (int k0 = 0; k0 < C_DIM; k0 += 32) {
        FragBF a[4], b[2];
#pragma unroll
        for (int mi = 0; mi < 4; ++mi) {
            a[mi].h[0] = *(const v8bf*)(ap[mi] + k0);
            a[mi].h[1] = *(const v8bf*)(ap[mi] + k0 + 16);
        }
#pragma unroll
        for (int ni = 0; ni < 2; ++ni) {
            b[ni].h[0] = *(const v8bf*)(bp[ni] + k0);
            b[ni].h[1] = *(const v8bf*)(bp[ni] + k0 + 16);
        }
#pragma unroll
        for (int mi = 0; mi < 4; ++mi)
#pragma unroll
            for (int ni = 0; ni < 2; ++ni)
                acc[mi][ni] = __builtin_amdgcn_wmma_f32_16x16x32_bf16(
                    false, a[mi].v, false, b[ni].v,
                    (short)0, acc[mi][ni], false, false);
    }

    if constexpr (!STORE_F32) {
        // Transposed bf16 store: lane owns column n, 8 contiguous rows ->
        // one packed 16B store per tile per lane.
        bf16_t* ob = outT + (size_t)bidx * NVOX * C_DIM;
#pragma unroll
        for (int mi = 0; mi < 4; ++mi) {
            const int mb = m0 + mi * 16 + 8 * h;
#pragma unroll
            for (int ni = 0; ni < 2; ++ni) {
                const int nn = n0 + ni * 16 + l15;
                PackBF t;
#pragma unroll
                for (int r = 0; r < 8; ++r)
                    t.e[r] = (__bf16)(acc[mi][ni][r] + bias[mb + r]);
                *(v8bf*)(ob + (size_t)nn * C_DIM + mb) = t.v;
            }
        }
    } else {
        float* of = outN + (size_t)bidx * C_DIM * (size_t)NVOX;
#pragma unroll
        for (int mi = 0; mi < 4; ++mi) {
            const int mb = m0 + mi * 16 + 8 * h;
#pragma unroll
            for (int ni = 0; ni < 2; ++ni) {
                const int nn = n0 + ni * 16 + l15;
#pragma unroll
                for (int r = 0; r < 8; ++r)
                    of[(size_t)(mb + r) * NVOX + nn] = acc[mi][ni][r] + bias[mb + r];
            }
        }
    }
}

// ---------------------------------------------------------------------------
// 4. scores[b,hq,hk,x,y] = (1/sqrt(32)) * sum_{p,z} qT[b,xy*48+z, hq*32+p]
//                                                  * kT[b,xy*48+z, hk*32+p]
// grid (48*48, BATCH), block 256. 4 threads per (hq,hk) pair.
// ---------------------------------------------------------------------------
__global__ __launch_bounds__(256) void scores_kernel(
    const bf16_t* __restrict__ qT, const bf16_t* __restrict__ kT,
    float* __restrict__ scores)
{
    __shared__ unsigned int qrow_u[128];
    __shared__ unsigned int krow_u[128];
    const int xy  = blockIdx.x;   // 0..2303
    const int b   = blockIdx.y;
    const int tid = threadIdx.x;
    const int pair = tid >> 2;    // 0..63 = hq*8+hk
    const int sub  = tid & 3;     // quarter of P range
    const int hq = pair >> 3, hk = pair & 7;

    const bf16_t* qb = qT + ((size_t)b * NVOX + (size_t)xy * 48) * C_DIM;
    const bf16_t* kb = kT + ((size_t)b * NVOX + (size_t)xy * 48) * C_DIM;

    float s = 0.f;
    for (int z = 0; z < 48; ++z) {
        __syncthreads();
        if (tid < 128)
            qrow_u[tid] = ((const unsigned int*)(qb + (size_t)z * C_DIM))[tid];
        else
            krow_u[tid - 128] =
                ((const unsigned int*)(kb + (size_t)z * C_DIM))[tid - 128];
        __syncthreads();
        const bf16_t* qh = ((const bf16_t*)qrow_u) + hq * 32 + sub * 8;
        const bf16_t* kh = ((const bf16_t*)krow_u) + hk * 32 + sub * 8;
#pragma unroll
        for (int p = 0; p < 8; ++p)
            s += (float)qh[p] * (float)kh[p];
    }
    s += __shfl_xor(s, 1, 32);
    s += __shfl_xor(s, 2, 32);
    if (sub == 0) {
        const int x = xy / 48, y = xy % 48;
        scores[(((size_t)b * 64 + pair) * 48 + x) * 48 + y] =
            s * 0.17677669529663687f;  // 1/sqrt(32)
    }
}

// ---------------------------------------------------------------------------
// 5. softmax over last axis (y, length 48), in place. One thread per row.
// ---------------------------------------------------------------------------
__global__ __launch_bounds__(256) void softmax_rows_kernel(
    float* __restrict__ scores, int nrows)
{
    const int row = blockIdx.x * blockDim.x + threadIdx.x;
    if (row >= nrows) return;
    float* r = scores + (size_t)row * 48;
    float mx = -3.0e38f;
#pragma unroll
    for (int y = 0; y < 48; ++y) mx = fmaxf(mx, r[y]);
    float e[48];
    float sum = 0.f;
#pragma unroll
    for (int y = 0; y < 48; ++y) { e[y] = __expf(r[y] - mx); sum += e[y]; }
    const float inv = 1.f / sum;
#pragma unroll
    for (int y = 0; y < 48; ++y) r[y] = e[y] * inv;
}

// ---------------------------------------------------------------------------
// 6. oT[b, n, h*32+c] = sum_m attn[b,h,m,x,y] * vT[b, n, m*32+c]
// grid (48*48, BATCH), block 256 (thread t owns output channel t).
// ---------------------------------------------------------------------------
__global__ __launch_bounds__(256) void attn_apply_kernel(
    const float* __restrict__ attn, const bf16_t* __restrict__ vT,
    bf16_t* __restrict__ oT)
{
    __shared__ float aw[64];
    __shared__ unsigned int vrow_u[128];
    const int xy  = blockIdx.x;
    const int b   = blockIdx.y;
    const int tid = threadIdx.x;
    const int x = xy / 48, y = xy % 48;

    if (tid < 64)
        aw[tid] = attn[(((size_t)b * 64 + tid) * 48 + x) * 48 + y];
    __syncthreads();

    const int n = tid >> 5;   // output head
    const int c = tid & 31;   // per-head channel
    float w[8];
#pragma unroll
    for (int m = 0; m < 8; ++m) w[m] = aw[n * 8 + m];

    const bf16_t* vb = vT + ((size_t)b * NVOX + (size_t)xy * 48) * C_DIM;
    bf16_t*       ob = oT + ((size_t)b * NVOX + (size_t)xy * 48) * C_DIM;

    for (int z = 0; z < 48; ++z) {
        __syncthreads();
        if (tid < 128)
            vrow_u[tid] = ((const unsigned int*)(vb + (size_t)z * C_DIM))[tid];
        __syncthreads();
        const bf16_t* vr = (const bf16_t*)vrow_u;
        float s = 0.f;
#pragma unroll
        for (int m = 0; m < 8; ++m)
            s += w[m] * (float)vr[m * 32 + c];
        ob[(size_t)z * C_DIM + tid] = (__bf16)s;
    }
}

// ---------------------------------------------------------------------------
// host launch
// ---------------------------------------------------------------------------
extern "C" void kernel_launch(void* const* d_in, const int* in_sizes, int n_in,
                              void* d_out, int out_size, void* d_ws, size_t ws_size,
                              hipStream_t stream)
{
    const float* x  = (const float*)d_in[0];
    const float* wq = (const float*)d_in[1];
    const float* bq = (const float*)d_in[2];
    const float* wk = (const float*)d_in[3];
    const float* bk = (const float*)d_in[4];
    const float* wv = (const float*)d_in[5];
    const float* bv = (const float*)d_in[6];
    const float* wo = (const float*)d_in[7];
    const float* bo = (const float*)d_in[8];
    float* out = (float*)d_out;
    (void)in_sizes; (void)n_in; (void)out_size; (void)ws_size;

    const size_t TBUF = (size_t)BATCH * NVOX * C_DIM;  // elements per buffer

    char* p = (char*)d_ws;
    bf16_t* xT  = (bf16_t*)p; p += TBUF * 2;
    bf16_t* qT  = (bf16_t*)p; p += TBUF * 2;
    bf16_t* kT  = (bf16_t*)p; p += TBUF * 2;
    bf16_t* vT  = (bf16_t*)p; p += TBUF * 2;
    bf16_t* oT  = (bf16_t*)p; p += TBUF * 2;
    bf16_t* wqb = (bf16_t*)p; p += 65536 * 2;
    bf16_t* wkb = (bf16_t*)p; p += 65536 * 2;
    bf16_t* wvb = (bf16_t*)p; p += 65536 * 2;
    bf16_t* wob = (bf16_t*)p; p += 65536 * 2;
    float* scores = (float*)p; p += (size_t)BATCH * 64 * 48 * 48 * 4;

    // 1. weight conversion
    cvt_f32_bf16_kernel<<<256, 256, 0, stream>>>(wq, wqb, 65536);
    cvt_f32_bf16_kernel<<<256, 256, 0, stream>>>(wk, wkb, 65536);
    cvt_f32_bf16_kernel<<<256, 256, 0, stream>>>(wv, wvb, 65536);
    cvt_f32_bf16_kernel<<<256, 256, 0, stream>>>(wo, wob, 65536);

    // 2. transpose x to voxel-major bf16
    transpose_to_bf16_kernel<<<dim3(NVOX / 32, C_DIM / 32, BATCH),
                               dim3(32, 8), 0, stream>>>(x, xT);

    // 3. q/k/v projections (WMMA), transposed bf16 outputs
    const dim3 ggrid(NVOX / 128, C_DIM / 128, BATCH);
    gemm256_wmma_kernel<false><<<ggrid, 256, 0, stream>>>(wqb, xT, bq, qT, nullptr);
    gemm256_wmma_kernel<false><<<ggrid, 256, 0, stream>>>(wkb, xT, bk, kT, nullptr);
    gemm256_wmma_kernel<false><<<ggrid, 256, 0, stream>>>(wvb, xT, bv, vT, nullptr);

    // 4. scores
    scores_kernel<<<dim3(48 * 48, BATCH), 256, 0, stream>>>(qT, kT, scores);

    // 5. softmax over y (in place)
    softmax_rows_kernel<<<(BATCH * 64 * 48 + 255) / 256, 256, 0, stream>>>(
        scores, BATCH * 64 * 48);

    // 6. apply attention over heads
    attn_apply_kernel<<<dim3(48 * 48, BATCH), 256, 0, stream>>>(scores, vT, oT);

    // 7. final o-projection, fp32 output in reference layout
    gemm256_wmma_kernel<true><<<ggrid, 256, 0, stream>>>(wob, oT, bo, oT, out);
}
